// GSN_61314953117896
// MI455X (gfx1250) — matrix operand order repaired and checked by reading, not compiled, not run on hardware
//
#include <hip/hip_runtime.h>
#include <hip/hip_bf16.h>

// ---------------------------------------------------------------------------
// GSN message-passing network on MI455X (gfx1250):
//  - all GEMMs via v_wmma_f32_16x16x32_f16 (f16 in, f32 accumulate)
//  - edge/node feature gathers via global_load_async_to_lds_b128 (ASYNCcnt)
//  - segment sums via global f32 atomics
// ---------------------------------------------------------------------------

typedef _Float16 h16_t;
typedef __attribute__((ext_vector_type(16))) _Float16 v16h;
typedef __attribute__((ext_vector_type(8)))  _Float16 v8h;
typedef __attribute__((ext_vector_type(8)))  float    v8f;

#define N_NODES   100000
#define N_EDGES   1600000
#define HID       64
#define EDGE_DIM  16
#define SC_DIM    6
#define MSG_K     160   // 150 padded to multiple of 32
#define UPD_K     128

// LDS row strides (halfs), padded to dodge bank conflicts
#define AS_E 168   // edge A tile (K=160)
#define AS_N 136   // node A tile (K=128)
#define AS_I 72    // init A tile (K=64)
#define HS   72    // hidden tile  (K=64)

static __device__ __forceinline__ void wait_lds() {
  asm volatile("s_wait_dscnt 0" ::: "memory");
}
static __device__ __forceinline__ void wait_async() {
  asm volatile("s_wait_asynccnt 0" ::: "memory");
}

// Async copy of 16 bytes global -> LDS (no VGPR staging, tracked by ASYNCcnt).
// lds_addr is the wave-relative LDS byte offset (flat shared ptr truncated to
// 32 bits: hardware uses addr[31:0] as the LDS address).
static __device__ __forceinline__ void async_copy16(unsigned lds_addr,
                                                    const void* gaddr) {
  asm volatile("global_load_async_to_lds_b128 %0, %1, off"
               :
               : "v"(lds_addr), "v"(gaddr)
               : "memory");
}

// Build a 16x32 f16 A-fragment for one lane from an LDS row.
// Lane L: row = L&15, sub = L>>4; elems 0..15 cover K = kb + sub*16 + 0..15
// (contiguous order, matching the packed B fragments).
static __device__ __forceinline__ v16h frag_a(const h16_t* rowbase, int kb, int sub) {
  v8h lo = *(const v8h*)(rowbase + kb + sub * 16);
  v8h hi = *(const v8h*)(rowbase + kb + sub * 16 + 8);
  return __builtin_shufflevector(lo, hi, 0,1,2,3,4,5,6,7,8,9,10,11,12,13,14,15);
}

static __device__ __forceinline__ v8f wmma16(v16h a, v16h b, v8f c) {
  return __builtin_amdgcn_wmma_f32_16x16x32_f16(false, a, false, b, (short)0, c,
                                                false, false);
}

// ---------------------------------------------------------------------------
// Pack a K x 64 f32 weight matrix into per-(Ktile, Ntile) WMMA B fragments:
// out[((t*4+n)*32 + L)*16 + e] = W[32t + 16*(L>>4) + e][16n + (L&15)], zero-pad
// rows >= K. Contiguous 32B v16h loads per lane in the GEMM kernels.
// ---------------------------------------------------------------------------
__global__ void gsn_pack_kernel(const float* __restrict__ W, h16_t* __restrict__ out,
                                int K, int Ktiles) {
  int total = Ktiles * 4 * 512;
  for (int p = blockIdx.x * blockDim.x + threadIdx.x; p < total;
       p += gridDim.x * blockDim.x) {
    int f = p >> 9, r = p & 511;
    int L = r >> 4, e = r & 15;
    int t = f >> 2, n = f & 3;
    int k = 32 * t + 16 * (L >> 4) + e;
    int c = 16 * n + (L & 15);
    float v = (k < K) ? W[k * 64 + c] : 0.0f;
    out[p] = (h16_t)v;
  }
}

__global__ void gsn_zero_kernel(float* __restrict__ p, long n) {
  long i  = (long)blockIdx.x * blockDim.x + threadIdx.x;
  long st = (long)gridDim.x * blockDim.x;
  for (; i < n; i += st) p[i] = 0.0f;
}

// ---------------------------------------------------------------------------
// Initial linear: h = x @ lin_w + lin_b  (no relu), stored as f16.
// One wave per 16-node tile.
// ---------------------------------------------------------------------------
__global__ __launch_bounds__(256) void gsn_init_kernel(
    const float* __restrict__ x, const h16_t* __restrict__ Blin,
    const float* __restrict__ lin_b, h16_t* __restrict__ h16, int n_tiles) {
  __shared__ h16_t Atile[8][16 * AS_I];
  const int w = threadIdx.x >> 5, L = threadIdx.x & 31;
  const int tile = blockIdx.x * 8 + w;
  if (tile >= n_tiles) return;
  const int row = L & 15, sub = L >> 4;
  const long node = (long)tile * 16 + row;
  h16_t* Ar = &Atile[w][row * AS_I];
#pragma unroll
  for (int i = 0; i < 32; ++i) {          // f32 -> f16 conversion pass
    int c = sub * 32 + i;
    Ar[c] = (h16_t)x[node * 64 + c];
  }
  wait_lds();
  v8f acc[4];
#pragma unroll
  for (int n = 0; n < 4; ++n)
#pragma unroll
    for (int i = 0; i < 8; ++i) acc[n][i] = 0.0f;
#pragma unroll
  for (int t = 0; t < 2; ++t) {
    v16h a = frag_a(Ar, 32 * t, sub);
#pragma unroll
    for (int n = 0; n < 4; ++n) {
      v16h b = *(const v16h*)(Blin + ((t * 4 + n) * 32 + L) * 16);
      acc[n] = wmma16(a, b, acc[n]);
    }
  }
#pragma unroll
  for (int n = 0; n < 4; ++n) {
    int c = 16 * n + row;
    float bc = lin_b[c];
#pragma unroll
    for (int v = 0; v < 8; ++v) {
      long nd = (long)tile * 16 + v + 8 * sub;
      h16[nd * 64 + c] = (h16_t)(acc[n][v] + bc);
    }
  }
}

// ---------------------------------------------------------------------------
// Edge MLP + scatter: one wave per 16-edge tile.
// A = [h[src] | h[dst] | sc[src] | sc[dst] | edge_feat | 0pad]  (16 x 160 f16)
// h rows gathered with async global->LDS copies; m = relu(A@W1+b1)@W2+b2;
// atomicAdd(upd[dst], m * ew).
// ---------------------------------------------------------------------------
__global__ __launch_bounds__(256) void gsn_edge_kernel(
    const h16_t* __restrict__ h16, const int* __restrict__ edge_list,
    const int* __restrict__ node_struct, const float* __restrict__ edge_feature,
    const float* __restrict__ edge_weight, const h16_t* __restrict__ Bw1,
    const float* __restrict__ b1, const h16_t* __restrict__ Bw2,
    const float* __restrict__ b2, float* __restrict__ upd, int n_tiles) {
  __shared__ h16_t Atile[8][16 * AS_E];
  __shared__ h16_t Htile[8][16 * HS];
  const int w = threadIdx.x >> 5, L = threadIdx.x & 31;
  const int tile = blockIdx.x * 8 + w;
  if (tile >= n_tiles) return;
  const int row = L & 15, sub = L >> 4;
  const int e0 = tile * 16;
  const int e  = e0 + row;
  const int src = edge_list[2 * e + 0];
  const int dst = edge_list[2 * e + 1];
  h16_t* Ar = &Atile[w][row * AS_E];
  const unsigned Ar_lds = (unsigned)(uintptr_t)Ar;   // LDS byte offset
  const h16_t* hs = h16 + (long)src * 64;
  const h16_t* hd = h16 + (long)dst * 64;
  // cols [0,64): h[src], cols [64,128): h[dst] — async global->LDS, 16B each
#pragma unroll
  for (int j = 0; j < 2; ++j) {
    int part = sub * 2 + j;
    async_copy16(Ar_lds + part * 32,       hs + part * 16);
    async_copy16(Ar_lds + 128 + part * 32, hd + part * 16);
  }
  // cols [128,160): sc_src(6) | sc_dst(6) | edge_feature(16) | 0pad(4)
#pragma unroll
  for (int i = 0; i < 16; ++i) {
    int c = 128 + sub * 16 + i;
    float v;
    if (c < 134)      v = (float)node_struct[src * 6 + (c - 128)];
    else if (c < 140) v = (float)node_struct[dst * 6 + (c - 134)];
    else if (c < 156) v = edge_feature[(long)e * 16 + (c - 140)];
    else              v = 0.0f;
    Ar[c] = (h16_t)v;
  }
  wait_async();
  wait_lds();

  v8f acc[4];
#pragma unroll
  for (int n = 0; n < 4; ++n)
#pragma unroll
    for (int i = 0; i < 8; ++i) acc[n][i] = 0.0f;
#pragma unroll
  for (int t = 0; t < 5; ++t) {           // K = 160
    v16h a = frag_a(Ar, 32 * t, sub);
#pragma unroll
    for (int n = 0; n < 4; ++n) {
      v16h b = *(const v16h*)(Bw1 + ((t * 4 + n) * 32 + L) * 16);
      acc[n] = wmma16(a, b, acc[n]);
    }
  }
#pragma unroll
  for (int n = 0; n < 4; ++n) {           // bias + relu -> hidden tile
    int c = 16 * n + row;
    float bc = b1[c];
#pragma unroll
    for (int v = 0; v < 8; ++v) {
      float val = acc[n][v] + bc;
      val = val > 0.0f ? val : 0.0f;
      Htile[w][(v + 8 * sub) * HS + c] = (h16_t)val;
    }
  }
  wait_lds();

  v8f acc2[4];
#pragma unroll
  for (int n = 0; n < 4; ++n)
#pragma unroll
    for (int i = 0; i < 8; ++i) acc2[n][i] = 0.0f;
  const h16_t* Hr = &Htile[w][row * HS];
#pragma unroll
  for (int t = 0; t < 2; ++t) {           // K = 64
    v16h a = frag_a(Hr, 32 * t, sub);
#pragma unroll
    for (int n = 0; n < 4; ++n) {
      v16h b = *(const v16h*)(Bw2 + ((t * 4 + n) * 32 + L) * 16);
      acc2[n] = wmma16(a, b, acc2[n]);
    }
  }
  int   dste[8];
  float ewe[8];
#pragma unroll
  for (int v = 0; v < 8; ++v) {
    int er = e0 + v + 8 * sub;
    dste[v] = edge_list[2 * er + 1];
    ewe[v]  = edge_weight[er];
  }
#pragma unroll
  for (int n = 0; n < 4; ++n) {
    int c = 16 * n + row;
    float bc = b2[c];
#pragma unroll
    for (int v = 0; v < 8; ++v) {
      float val = (acc2[n][v] + bc) * ewe[v];
      atomicAdd(&upd[(long)dste[v] * 64 + c], val);   // segment_sum over dst
    }
  }
}

// ---------------------------------------------------------------------------
// Node update MLP: z = relu([h|upd] @ W1 + b1) @ W2 + b2 ; h = relu(z).
// Last layer also writes f32 node_feature and atomically accumulates
// graph_feature = segment_sum(h, node2graph).
// ---------------------------------------------------------------------------
__global__ __launch_bounds__(256) void gsn_node_kernel(
    const h16_t* __restrict__ h16_in, const float* __restrict__ upd,
    const h16_t* __restrict__ Bw1, const float* __restrict__ b1,
    const h16_t* __restrict__ Bw2, const float* __restrict__ b2,
    h16_t* __restrict__ h16_out, float* __restrict__ out_node,
    float* __restrict__ out_graph, const int* __restrict__ node2graph,
    int write_out, int n_tiles) {
  __shared__ h16_t Atile[8][16 * AS_N];
  __shared__ h16_t Htile[8][16 * HS];
  const int w = threadIdx.x >> 5, L = threadIdx.x & 31;
  const int tile = blockIdx.x * 8 + w;
  if (tile >= n_tiles) return;
  const int row = L & 15, sub = L >> 4;
  const long node = (long)tile * 16 + row;
  h16_t* Ar = &Atile[w][row * AS_N];
  const unsigned Ar_lds = (unsigned)(uintptr_t)Ar;
#pragma unroll
  for (int j = 0; j < 2; ++j) {           // cols 0..63 : h16, async copies
    int part = sub * 2 + j;
    async_copy16(Ar_lds + part * 32, h16_in + node * 64 + part * 16);
  }
#pragma unroll
  for (int i = 0; i < 32; ++i) {          // cols 64..127 : upd (f32 -> f16)
    int c = 64 + sub * 32 + i;
    Ar[c] = (h16_t)upd[node * 64 + (c - 64)];
  }
  wait_async();
  wait_lds();

  v8f acc[4];
#pragma unroll
  for (int n = 0; n < 4; ++n)
#pragma unroll
    for (int i = 0; i < 8; ++i) acc[n][i] = 0.0f;
#pragma unroll
  for (int t = 0; t < 4; ++t) {           // K = 128
    v16h a = frag_a(Ar, 32 * t, sub);
#pragma unroll
    for (int n = 0; n < 4; ++n) {
      v16h b = *(const v16h*)(Bw1 + ((t * 4 + n) * 32 + L) * 16);
      acc[n] = wmma16(a, b, acc[n]);
    }
  }
#pragma unroll
  for (int n = 0; n < 4; ++n) {
    int c = 16 * n + row;
    float bc = b1[c];
#pragma unroll
    for (int v = 0; v < 8; ++v) {
      float val = acc[n][v] + bc;
      val = val > 0.0f ? val : 0.0f;
      Htile[w][(v + 8 * sub) * HS + c] = (h16_t)val;
    }
  }
  wait_lds();

  v8f acc2[4];
#pragma unroll
  for (int n = 0; n < 4; ++n)
#pragma unroll
    for (int i = 0; i < 8; ++i) acc2[n][i] = 0.0f;
  const h16_t* Hr = &Htile[w][row * HS];
#pragma unroll
  for (int t = 0; t < 2; ++t) {           // K = 64
    v16h a = frag_a(Hr, 32 * t, sub);
#pragma unroll
    for (int n = 0; n < 4; ++n) {
      v16h b = *(const v16h*)(Bw2 + ((t * 4 + n) * 32 + L) * 16);
      acc2[n] = wmma16(a, b, acc2[n]);
    }
  }
  int n2g[8];
  if (write_out) {
#pragma unroll
    for (int v = 0; v < 8; ++v) n2g[v] = node2graph[tile * 16 + v + 8 * sub];
  }
#pragma unroll
  for (int n = 0; n < 4; ++n) {
    int c = 16 * n + row;
    float bc = b2[c];
#pragma unroll
    for (int v = 0; v < 8; ++v) {
      long nd = (long)tile * 16 + v + 8 * sub;
      float val = acc2[n][v] + bc;
      val = val > 0.0f ? val : 0.0f;    // h = relu(z)
      h16_out[nd * 64 + c] = (h16_t)val;
      if (write_out) {
        out_node[nd * 64 + c] = val;
        atomicAdd(&out_graph[n2g[v] * 64 + c], val);
      }
    }
  }
}

// ---------------------------------------------------------------------------
extern "C" void kernel_launch(void* const* d_in, const int* in_sizes, int n_in,
                              void* d_out, int out_size, void* d_ws, size_t ws_size,
                              hipStream_t stream) {
  const float* x            = (const float*)d_in[0];
  const int*   edge_list    = (const int*)d_in[1];
  const int*   node_struct  = (const int*)d_in[2];
  const float* edge_feature = (const float*)d_in[3];
  const float* edge_weight  = (const float*)d_in[4];
  const int*   node2graph   = (const int*)d_in[5];
  const float* lin_w        = (const float*)d_in[7];
  const float* lin_b        = (const float*)d_in[8];
  const float* msg_w1       = (const float*)d_in[9];
  const float* msg_b1       = (const float*)d_in[10];
  const float* msg_w2       = (const float*)d_in[11];
  const float* msg_b2       = (const float*)d_in[12];
  const float* upd_w1       = (const float*)d_in[13];
  const float* upd_b1       = (const float*)d_in[14];
  const float* upd_w2       = (const float*)d_in[15];
  const float* upd_b2       = (const float*)d_in[16];
  (void)in_sizes; (void)n_in; (void)out_size; (void)ws_size;

  float* out       = (float*)d_out;
  float* out_graph = out;                 // 128*64
  float* out_node  = out + 128 * 64;      // 100000*64

  char* ws = (char*)d_ws;
  h16_t* h16    = (h16_t*)(ws);                        // 12.8 MB
  float* upd    = (float*)(ws + 12800000);             // 25.6 MB
  h16_t* packed = (h16_t*)(ws + 38400000);             // ~115 KB
  h16_t* lin_p  = packed;                              // 2 Ktiles  -> 4096 halfs
  h16_t* mw1_p  = lin_p + 4096;                        // 5 Ktiles  -> 10240 * 2
  h16_t* mw2_p  = mw1_p + 2 * 10240;                   // 2 Ktiles  -> 4096 * 2
  h16_t* uw1_p  = mw2_p + 2 * 4096;                    // 4 Ktiles  -> 8192 * 2
  h16_t* uw2_p  = uw1_p + 2 * 8192;                    // 2 Ktiles  -> 4096 * 2

  // Pack all weight matrices into WMMA B-fragment layout (f16).
  gsn_pack_kernel<<<16, 256, 0, stream>>>(lin_w, lin_p, 64, 2);
  for (int l = 0; l < 2; ++l) {
    gsn_pack_kernel<<<40, 256, 0, stream>>>(msg_w1 + l * 150 * 64, mw1_p + l * 10240, 150, 5);
    gsn_pack_kernel<<<16, 256, 0, stream>>>(msg_w2 + l * 64 * 64,  mw2_p + l * 4096,  64, 2);
    gsn_pack_kernel<<<32, 256, 0, stream>>>(upd_w1 + l * 128 * 64, uw1_p + l * 8192, 128, 4);
    gsn_pack_kernel<<<16, 256, 0, stream>>>(upd_w2 + l * 64 * 64,  uw2_p + l * 4096,  64, 2);
  }
  gsn_zero_kernel<<<32, 256, 0, stream>>>(out_graph, 128 * 64);

  const int node_tiles = N_NODES / 16;   // 6250
  const int edge_tiles = N_EDGES / 16;   // 100000
  const int node_blocks = (node_tiles + 7) / 8;
  const int edge_blocks = (edge_tiles + 7) / 8;

  gsn_init_kernel<<<node_blocks, 256, 0, stream>>>(x, lin_p, lin_b, h16, node_tiles);

  for (int l = 0; l < 2; ++l) {
    gsn_zero_kernel<<<4096, 256, 0, stream>>>(upd, (long)N_NODES * 64);
    gsn_edge_kernel<<<edge_blocks, 256, 0, stream>>>(
        h16, edge_list, node_struct, edge_feature, edge_weight,
        mw1_p + l * 10240, msg_b1 + l * 64, mw2_p + l * 4096, msg_b2 + l * 64,
        upd, edge_tiles);
    gsn_node_kernel<<<node_blocks, 256, 0, stream>>>(
        h16, upd, uw1_p + l * 8192, upd_b1 + l * 64, uw2_p + l * 4096,
        upd_b2 + l * 64, h16, out_node, out_graph, node2graph,
        (l == 1) ? 1 : 0, node_tiles);
  }
}
